// BlockGT_10608569221420
// MI455X (gfx1250) — matrix operand order repaired and checked by reading, not compiled, not run on hardware
//
#include <hip/hip_runtime.h>
#include <cstdint>

// ---------------------------------------------------------------------------
// Problem constants
// ---------------------------------------------------------------------------
#define Bq   2
#define Nn   192
#define Dd   256
#define Hh   8
#define DHd  32
#define Dff  1024
#define Me   (Bq * Nn * Nn)        // 73728 edge rows
#define Mx   (Bq * Nn)             // 384 node rows
static constexpr float SCALE = 0.17677669529663687f; // 1/sqrt(32)

typedef __bf16 v16bf __attribute__((ext_vector_type(16)));
typedef float  v8f   __attribute__((ext_vector_type(8)));

union FragB { unsigned u[8]; v16bf v; };
union FragC { v8f v; float f[8]; };

// ---------------------------------------------------------------------------
// bf16 helpers (raw ushort storage, RNE convert)
// ---------------------------------------------------------------------------
__device__ __forceinline__ unsigned short f2bf(float f) {
    union { float f; unsigned u; } v; v.f = f;
    unsigned r = v.u + 0x7FFFu + ((v.u >> 16) & 1u);
    return (unsigned short)(r >> 16);
}
__device__ __forceinline__ float bf2f(unsigned short h) {
    union { unsigned u; float f; } v; v.u = ((unsigned)h) << 16; return v.f;
}

// ---------------------------------------------------------------------------
// CDNA5 async global->LDS copy (ASYNCcnt-tracked, bypasses VGPRs).
// Generic LDS pointer low 32 bits == LDS byte offset (ISA 10.2 aperture rule).
// ---------------------------------------------------------------------------
__device__ __forceinline__ void async_b128(void* lds, const void* gptr) {
    unsigned off = (unsigned)(uintptr_t)lds;
    asm volatile("global_load_async_to_lds_b128 %0, %1, off"
                 :: "v"(off), "v"(gptr) : "memory");
}
__device__ __forceinline__ void wait_async0() {
    asm volatile("s_wait_asynccnt 0" ::: "memory");
}

// ---------------------------------------------------------------------------
// WMMA fragment helpers — layouts per CDNA5 ISA 7.12.2
// A (16x32 bf16): lane half h=lane>>4, m=lane&15; vgpr p holds K = {2p | 2p+8(p>=4)} + 8h
// B (32x16 bf16): n=lane&15; vgpr p holds K = {2p | 2p+8(p>=4)} + 8h (same as A)
// C/D (16x16 f32): vgpr i -> M = i + 8*(lane>>4), N = lane&15
// ---------------------------------------------------------------------------
__device__ __forceinline__ void load_afrag(const unsigned short* base, int ldk,
                                           int lane, unsigned* au) {
    const int half = lane >> 4, m = lane & 15;
    const unsigned short* r = base + m * ldk + half * 8;
#pragma unroll
    for (int p = 0; p < 8; ++p) {
        int k = (p < 4) ? (2 * p) : (2 * p + 8);
        au[p] = *(const unsigned*)(r + k);
    }
}
__device__ __forceinline__ void load_bfrag(const unsigned short* base, int ldn,
                                           int lane, unsigned* bu) {
    const unsigned short* r = base + (lane & 15) + (lane >> 4) * 8 * ldn;
#pragma unroll
    for (int p = 0; p < 8; ++p) {
        int k = (p < 4) ? (2 * p) : (2 * p + 8);
        bu[p] = (unsigned)r[k * ldn] | ((unsigned)r[(k + 1) * ldn] << 16);
    }
}
__device__ __forceinline__ v8f wmma_bf16(const unsigned* a, const unsigned* b, v8f c) {
    FragB A, B;
#pragma unroll
    for (int i = 0; i < 8; ++i) { A.u[i] = a[i]; B.u[i] = b[i]; }
    return __builtin_amdgcn_wmma_f32_16x16x32_bf16(false, A.v, false, B.v,
                                                   (short)0, c, false, false);
}
__device__ __forceinline__ float wsum32(float v) {
#pragma unroll
    for (int o = 16; o > 0; o >>= 1) v += __shfl_xor(v, o, 32);
    return v;
}

// ---------------------------------------------------------------------------
// LayerNorm over rows of 256. Writes fp32 copy (residual) and/or bf16 copy.
// ---------------------------------------------------------------------------
__global__ __launch_bounds__(256) void ln_kernel(const float* __restrict__ in,
                                                 const float* __restrict__ g,
                                                 const float* __restrict__ b,
                                                 float* __restrict__ outf,
                                                 unsigned short* __restrict__ outb) {
    __shared__ float red[256];
    const long row = blockIdx.x;
    const int t = threadIdx.x;
    float v = in[row * Dd + t];
    red[t] = v; __syncthreads();
#pragma unroll
    for (int s = 128; s > 0; s >>= 1) { if (t < s) red[t] += red[t + s]; __syncthreads(); }
    float mean = red[0] * (1.f / Dd);
    __syncthreads();
    float d = v - mean;
    red[t] = d * d; __syncthreads();
#pragma unroll
    for (int s = 128; s > 0; s >>= 1) { if (t < s) red[t] += red[t + s]; __syncthreads(); }
    float var = red[0] * (1.f / Dd);
    float y = d * rsqrtf(var + 1e-5f) * g[t] + b[t];
    if (outf) outf[row * Dd + t] = y;
    if (outb) outb[row * Dd + t] = f2bf(y);
}

// ---------------------------------------------------------------------------
// Weight packing: fp32 -> bf16, transposed into k-major [Din x ld] matrices
// ---------------------------------------------------------------------------
__global__ void pack_heads(const float* __restrict__ W, // [H=8, 256, 32]
                           unsigned short* __restrict__ out, int ld, int colOff) {
    int idx = blockIdx.x * 256 + threadIdx.x;   // 65536 total
    int n = idx & 255, k = idx >> 8;
    out[k * ld + colOff + n] = f2bf(W[(n >> 5) * (Dd * DHd) + k * DHd + (n & 31)]);
}
__global__ void pack_dense(const float* __restrict__ W, // [Dout, Din] row-major
                           unsigned short* __restrict__ out, int Din, int Dout, int ld) {
    int idx = blockIdx.x * 256 + threadIdx.x;   // Din*Dout total
    int k = idx / Dout, n = idx % Dout;
    out[k * ld + n] = f2bf(W[n * Din + k]);
}
__global__ void pack_bias3(const float* __restrict__ a, const float* __restrict__ b,
                           const float* __restrict__ c, float* __restrict__ out) {
    int t = blockIdx.x * 256 + threadIdx.x;     // 768 total
    out[t] = (t < 256) ? a[t] : (t < 512 ? b[t - 256] : c[t - 512]);
}

// ---------------------------------------------------------------------------
// Tiled WMMA GEMM:  C[M x N] = A[M x K] * Bt[K x N] (+bias)
// BM=128, BN=64, BK=32; 8 waves, each owning a 16x64 strip (4 accum tiles).
// Double-buffered LDS tiles, staged with async global->LDS (ASYNCcnt).
// Mode: Cb != null -> write bf16;  else Cf += result (fp32 residual add).
// ---------------------------------------------------------------------------
__global__ __launch_bounds__(256) void gemm_bf16(
    const unsigned short* __restrict__ A, int lda,
    const unsigned short* __restrict__ Bt, int ldb,
    const float* __restrict__ bias,
    unsigned short* __restrict__ Cb, float* __restrict__ Cf, int ldc, int K) {
    __shared__ __align__(16) unsigned short As[2][128 * 32];
    __shared__ __align__(16) unsigned short Bs[2][32 * 64];
    const int tid = threadIdx.x, wave = tid >> 5, lane = tid & 31;
    const long rowBase = (long)blockIdx.x * 128;
    const int nBase = blockIdx.y * 64;

    // per-thread staging slots: A tile 128x32 (32B/thread), B tile 32x64 (16B/thread)
    const int rA = tid >> 1, cA = (tid & 1) * 16;
    const int rB = tid >> 3, cB = (tid & 7) * 8;
    const unsigned short* gA = A + (rowBase + rA) * lda + cA;
    const unsigned short* gB = Bt + (long)rB * ldb + nBase + cB;

    auto stage = [&](int buf, int k0) {
        async_b128(&As[buf][rA * 32 + cA],     gA + k0);
        async_b128(&As[buf][rA * 32 + cA + 8], gA + k0 + 8);
        async_b128(&Bs[buf][rB * 64 + cB],     gB + (long)k0 * ldb);
    };

    v8f acc0 = {}, acc1 = {}, acc2 = {}, acc3 = {};
    const int ksteps = K >> 5;
    stage(0, 0);
    for (int ks = 0; ks < ksteps; ++ks) {
        const int cur = ks & 1;
        wait_async0();          // this wave's async copies landed in LDS
        __syncthreads();        // everyone's copies visible
        if (ks + 1 < ksteps) stage(cur ^ 1, (ks + 1) << 5);  // overlap next tile

        unsigned au[8];
        load_afrag(&As[cur][(wave * 16) * 32], 32, lane, au);
        unsigned bu[8];
        load_bfrag(&Bs[cur][0],  64, lane, bu); acc0 = wmma_bf16(au, bu, acc0);
        load_bfrag(&Bs[cur][16], 64, lane, bu); acc1 = wmma_bf16(au, bu, acc1);
        load_bfrag(&Bs[cur][32], 64, lane, bu); acc2 = wmma_bf16(au, bu, acc2);
        load_bfrag(&Bs[cur][48], 64, lane, bu); acc3 = wmma_bf16(au, bu, acc3);
    }

    v8f accs[4] = {acc0, acc1, acc2, acc3};
#pragma unroll
    for (int nf = 0; nf < 4; ++nf) {
        FragC fc; fc.v = accs[nf];
#pragma unroll
        for (int i = 0; i < 8; ++i) {
            int m = ((lane >> 4) * 8) + i;
            int n = nBase + nf * 16 + (lane & 15);
            long row = rowBase + wave * 16 + m;
            float v = fc.f[i] + (bias ? bias[n] : 0.f);
            if (Cb) Cb[row * ldc + n] = f2bf(v);
            else    Cf[row * ldc + n] += v;
        }
    }
}

// ---------------------------------------------------------------------------
// Edge->node attention: node i attends over its N edges. One wave per (b,i,h),
// lane = head-dim element (DH == wave32).
// ---------------------------------------------------------------------------
__global__ __launch_bounds__(256) void attn_e2n(
    const unsigned short* __restrict__ xproj,  // [Mx x 768]: Q at col 0
    const unsigned short* __restrict__ eproj,  // [Me x 768]: Ke col 0, Ve col 256
    unsigned short* __restrict__ xheads) {     // [Mx x 256] bf16
    __shared__ float sbuf[8][Nn];
    const int wave = threadIdx.x >> 5, lane = threadIdx.x & 31;
    const int g = blockIdx.x * 8 + wave;       // 0..3071
    const int bi = g >> 3, h = g & 7;          // bi = b*N+i
    const float q = bf2f(xproj[bi * 768 + h * 32 + lane]);

    float mx = -1e30f;
    for (int j = 0; j < Nn; ++j) {
        float kv = bf2f(eproj[((long)bi * Nn + j) * 768 + h * 32 + lane]);
        float s = wsum32(q * kv) * SCALE;
        sbuf[wave][j] = s;
        mx = fmaxf(mx, s);
    }
    float den = 0.f, acc = 0.f;
    for (int j = 0; j < Nn; ++j) {
        float w = __expf(sbuf[wave][j] - mx);
        den += w;
        acc += w * bf2f(eproj[((long)bi * Nn + j) * 768 + 256 + h * 32 + lane]);
    }
    xheads[bi * Dd + h * 32 + lane] = f2bf(acc / den);
}

// ---------------------------------------------------------------------------
// Node->edge attention: edge (i,j) attends over endpoints. One block per edge,
// one wave per head, lane = head-dim element.
// ---------------------------------------------------------------------------
__global__ __launch_bounds__(256) void attn_n2e(
    const unsigned short* __restrict__ eproj,  // Qe at col 512
    const unsigned short* __restrict__ xproj,  // Kx col 256, Vx col 512
    unsigned short* __restrict__ eheads) {     // [Me x 256] bf16
    const int h = threadIdx.x >> 5, lane = threadIdx.x & 31;
    const long edge = blockIdx.x;
    const int b = (int)(edge / (Nn * Nn));
    const int rem = (int)(edge % (Nn * Nn));
    const int i = rem / Nn, j = rem % Nn;
    const int bi = b * Nn + i, bj = b * Nn + j;

    float qe  = bf2f(eproj[edge * 768 + 512 + h * 32 + lane]);
    float kxi = bf2f(xproj[bi * 768 + 256 + h * 32 + lane]);
    float kxj = bf2f(xproj[bj * 768 + 256 + h * 32 + lane]);
    float si = wsum32(qe * kxi) * SCALE;
    float sj = wsum32(qe * kxj) * SCALE;
    float m = fmaxf(si, sj);
    float ei = __expf(si - m), ej = __expf(sj - m);
    float inv = 1.f / (ei + ej);
    float vxi = bf2f(xproj[bi * 768 + 512 + h * 32 + lane]);
    float vxj = bf2f(xproj[bj * 768 + 512 + h * 32 + lane]);
    eheads[edge * Dd + h * 32 + lane] = f2bf((ei * vxi + ej * vxj) * inv);
}

// ---------------------------------------------------------------------------
// Fused MLP: Out[64x256] += W2( leaky(W1(H)+b1) ) + b2, hidden tile in LDS.
// 160 KB dynamic LDS (fits CDNA5's 320 KB WGP LDS). 512 WMMAs per wave.
// ---------------------------------------------------------------------------
__global__ __launch_bounds__(256) void mlp_fused(
    const unsigned short* __restrict__ Hin,  // [M x 256] bf16
    const unsigned short* __restrict__ W1t,  // [256 x 1024] k-major bf16
    const float* __restrict__ b1,
    const unsigned short* __restrict__ W2t,  // [1024 x 256] k-major bf16
    const float* __restrict__ b2,
    float* __restrict__ Out) {               // [M x 256] fp32, accumulated
    extern __shared__ __align__(16) unsigned short smem[];
    unsigned short* sH = smem;               // 64 x 256
    unsigned short* sG = smem + 64 * Dd;     // 64 x 1024
    const int tid = threadIdx.x, wave = tid >> 5, lane = tid & 31;
    const long rowBase = (long)blockIdx.x * 64;

#pragma unroll
    for (int i = 0; i < 8; ++i) {            // async-stage H tile (16B chunks)
        int c = tid + i * 256;
        async_b128(sH + c * 8, Hin + rowBase * Dd + c * 8);
    }
    wait_async0();
    __syncthreads();

    // stage 1: hidden = leaky(H @ W1t + b1), each wave owns 128 hidden cols
    for (int mt = 0; mt < 4; ++mt) {
        unsigned Af[8][8];
#pragma unroll
        for (int ks = 0; ks < 8; ++ks)
            load_afrag(sH + (mt * 16) * Dd + ks * 32, Dd, lane, Af[ks]);
        for (int nt = 0; nt < 8; ++nt) {
            const int n0 = wave * 128 + nt * 16;
            v8f acc = {};
#pragma unroll
            for (int ks = 0; ks < 8; ++ks) {
                unsigned bu[8];
                load_bfrag(W1t + (long)(ks * 32) * Dff + n0, Dff, lane, bu);
                acc = wmma_bf16(Af[ks], bu, acc);
            }
            FragC fc; fc.v = acc;
#pragma unroll
            for (int i = 0; i < 8; ++i) {
                int m = ((lane >> 4) * 8) + i;
                int n = n0 + (lane & 15);
                float v = fc.f[i] + b1[n];
                v = v > 0.f ? v : 0.01f * v;
                sG[(mt * 16 + m) * Dff + n] = f2bf(v);
            }
        }
    }
    __syncthreads();

    // stage 2: Out += hidden @ W2t + b2, each wave owns 32 output cols
    for (int mt = 0; mt < 4; ++mt) {
        for (int nt = 0; nt < 2; ++nt) {
            const int n0 = wave * 32 + nt * 16;
            v8f acc = {};
            for (int ks = 0; ks < 32; ++ks) {
                unsigned au[8], bu[8];
                load_afrag(sG + (mt * 16) * Dff + ks * 32, Dff, lane, au);
                load_bfrag(W2t + (long)(ks * 32) * Dd + n0, Dd, lane, bu);
                acc = wmma_bf16(au, bu, acc);
            }
            FragC fc; fc.v = acc;
#pragma unroll
            for (int i = 0; i < 8; ++i) {
                int m = ((lane >> 4) * 8) + i;
                int n = n0 + (lane & 15);
                Out[(rowBase + mt * 16 + m) * Dd + n] += fc.f[i] + b2[n];
            }
        }
    }
}

// ---------------------------------------------------------------------------
// Host orchestration
// ---------------------------------------------------------------------------
extern "C" void kernel_launch(void* const* d_in, const int* in_sizes, int n_in,
                              void* d_out, int out_size, void* d_ws, size_t ws_size,
                              hipStream_t stream) {
    (void)in_sizes; (void)n_in; (void)out_size; (void)ws_size;
    const float* x      = (const float*)d_in[0];
    const float* e      = (const float*)d_in[1];
    const float* ln_x_g = (const float*)d_in[2],  *ln_x_b = (const float*)d_in[3];
    const float* ln_e_g = (const float*)d_in[4],  *ln_e_b = (const float*)d_in[5];
    const float* ln_x2_g= (const float*)d_in[6],  *ln_x2_b= (const float*)d_in[7];
    const float* ln_e2_g= (const float*)d_in[8],  *ln_e2_b= (const float*)d_in[9];
    const float* Wq_en = (const float*)d_in[10], *bq_en = (const float*)d_in[11];
    const float* Wk_en = (const float*)d_in[12], *bk_en = (const float*)d_in[13];
    const float* Wv_en = (const float*)d_in[14], *bv_en = (const float*)d_in[15];
    const float* Wq_ne = (const float*)d_in[16], *bq_ne = (const float*)d_in[17];
    const float* Wk_ne = (const float*)d_in[18], *bk_ne = (const float*)d_in[19];
    const float* Wv_ne = (const float*)d_in[20], *bv_ne = (const float*)d_in[21];
    const float* WOx_w = (const float*)d_in[22], *WOx_b = (const float*)d_in[23];
    const float* WOe_w = (const float*)d_in[24], *WOe_b = (const float*)d_in[25];
    const float* w1x = (const float*)d_in[26], *b1x = (const float*)d_in[27];
    const float* w2x = (const float*)d_in[28], *b2x = (const float*)d_in[29];
    const float* w1e = (const float*)d_in[30], *b1e = (const float*)d_in[31];
    const float* w2e = (const float*)d_in[32], *b2e = (const float*)d_in[33];

    float* outx = (float*)d_out;
    float* oute = outx + (long)Mx * Dd;

    unsigned char* base = (unsigned char*)d_ws;
    size_t off = 0;
    auto alloc = [&](size_t bytes) -> void* {
        off = (off + 255) & ~(size_t)255;
        void* r = base + off; off += bytes; return r;
    };
    unsigned short* xnb    = (unsigned short*)alloc((size_t)Mx * Dd * 2);
    unsigned short* enb    = (unsigned short*)alloc((size_t)Me * Dd * 2);
    unsigned short* eproj  = (unsigned short*)alloc((size_t)Me * 768 * 2);
    unsigned short* xproj  = (unsigned short*)alloc((size_t)Mx * 768 * 2);
    unsigned short* xheads = (unsigned short*)alloc((size_t)Mx * Dd * 2);
    unsigned short* eheads = (unsigned short*)alloc((size_t)Me * Dd * 2);
    unsigned short* he_e   = (unsigned short*)alloc((size_t)Me * Dd * 2);
    unsigned short* he_x   = (unsigned short*)alloc((size_t)Mx * Dd * 2);
    unsigned short* Wpe    = (unsigned short*)alloc((size_t)Dd * 768 * 2);
    unsigned short* Wpx    = (unsigned short*)alloc((size_t)Dd * 768 * 2);
    unsigned short* WOxT   = (unsigned short*)alloc((size_t)Dd * Dd * 2);
    unsigned short* WOeT   = (unsigned short*)alloc((size_t)Dd * Dd * 2);
    unsigned short* W1xT   = (unsigned short*)alloc((size_t)Dd * Dff * 2);
    unsigned short* W2xT   = (unsigned short*)alloc((size_t)Dff * Dd * 2);
    unsigned short* W1eT   = (unsigned short*)alloc((size_t)Dd * Dff * 2);
    unsigned short* W2eT   = (unsigned short*)alloc((size_t)Dff * Dd * 2);
    float* bias_e = (float*)alloc(768 * 4);
    float* bias_x = (float*)alloc(768 * 4);

    // ---- weight / bias packing (fp32 -> bf16, transposed k-major) ----
    pack_heads<<<256, 256, 0, stream>>>(Wk_en, Wpe, 768, 0);
    pack_heads<<<256, 256, 0, stream>>>(Wv_en, Wpe, 768, 256);
    pack_heads<<<256, 256, 0, stream>>>(Wq_ne, Wpe, 768, 512);
    pack_heads<<<256, 256, 0, stream>>>(Wq_en, Wpx, 768, 0);
    pack_heads<<<256, 256, 0, stream>>>(Wk_ne, Wpx, 768, 256);
    pack_heads<<<256, 256, 0, stream>>>(Wv_ne, Wpx, 768, 512);
    pack_dense<<<256, 256, 0, stream>>>(WOx_w, WOxT, Dd, Dd, Dd);
    pack_dense<<<256, 256, 0, stream>>>(WOe_w, WOeT, Dd, Dd, Dd);
    pack_dense<<<1024, 256, 0, stream>>>(w1x, W1xT, Dd, Dff, Dff);
    pack_dense<<<1024, 256, 0, stream>>>(w2x, W2xT, Dff, Dd, Dd);
    pack_dense<<<1024, 256, 0, stream>>>(w1e, W1eT, Dd, Dff, Dff);
    pack_dense<<<1024, 256, 0, stream>>>(w2e, W2eT, Dff, Dd, Dd);
    pack_bias3<<<3, 256, 0, stream>>>(bk_en, bv_en, bq_ne, bias_e);
    pack_bias3<<<3, 256, 0, stream>>>(bq_en, bk_ne, bv_ne, bias_x);

    // ---- LN1: outx/oute get the fp32 LN copy (residual base), bf16 side copies ----
    ln_kernel<<<Mx, 256, 0, stream>>>(x, ln_x_g, ln_x_b, outx, xnb);
    ln_kernel<<<Me, 256, 0, stream>>>(e, ln_e_g, ln_e_b, oute, enb);

    // ---- fused QKV projections (bf16 WMMA GEMMs) ----
    gemm_bf16<<<dim3(Mx / 128, 12), 256, 0, stream>>>(xnb, Dd, Wpx, 768, bias_x,
                                                      xproj, nullptr, 768, Dd);
    gemm_bf16<<<dim3(Me / 128, 12), 256, 0, stream>>>(enb, Dd, Wpe, 768, bias_e,
                                                      eproj, nullptr, 768, Dd);

    // ---- attention ----
    attn_e2n<<<Mx * Hh / 8, 256, 0, stream>>>(xproj, eproj, xheads);
    attn_n2e<<<Me, 256, 0, stream>>>(eproj, xproj, eheads);

    // ---- output projections, accumulated onto fp32 residuals in d_out ----
    gemm_bf16<<<dim3(Mx / 128, 4), 256, 0, stream>>>(xheads, Dd, WOxT, Dd, WOx_b,
                                                     nullptr, outx, Dd, Dd);
    gemm_bf16<<<dim3(Me / 128, 4), 256, 0, stream>>>(eheads, Dd, WOeT, Dd, WOe_b,
                                                     nullptr, oute, Dd, Dd);

    // ---- LN2 -> bf16 MLP inputs ----
    ln_kernel<<<Mx, 256, 0, stream>>>(outx, ln_x2_g, ln_x2_b, nullptr, he_x);
    ln_kernel<<<Me, 256, 0, stream>>>(oute, ln_e2_g, ln_e2_b, nullptr, he_e);

    // ---- fused MLPs (hidden tile resident in 160 KB LDS) ----
    const int mlp_lds = (64 * Dd + 64 * Dff) * 2; // 163840 bytes
    hipFuncSetAttribute(reinterpret_cast<const void*>(mlp_fused),
                        hipFuncAttributeMaxDynamicSharedMemorySize, mlp_lds);
    mlp_fused<<<Mx / 64, 256, mlp_lds, stream>>>(he_x, W1xT, b1x, W2xT, b2x, outx);
    mlp_fused<<<Me / 64, 256, mlp_lds, stream>>>(he_e, W1eT, b1e, W2eT, b2e, oute);
}
